// CausalSlidingWindowAttention_85392539779397
// MI455X (gfx1250) — matrix operand order repaired and checked by reading, not compiled, not run on hardware
//
#include <hip/hip_runtime.h>
#include <hip/hip_bf16.h>

typedef __attribute__((ext_vector_type(16))) _Float16 v16h;
typedef __attribute__((ext_vector_type(8)))  _Float16 v8h;
typedef __attribute__((ext_vector_type(8)))  float    v8f;

constexpr int B_  = 2;
constexpr int T_  = 2048;
constexpr int D_  = 1024;
constexpr int H_  = 16;
constexpr int HD_ = 64;
constexpr int WIN = 128;
constexpr int TPAD = T_ + 192;          // zero-padded key dim for V^T loads
constexpr float NEGINF = -1e30f;

// ---------------------------------------------------------------------------
// WMMA helpers (CDNA5 wave32, V_WMMA_F32_16X16X32_F16)
// ---------------------------------------------------------------------------
__device__ __forceinline__ v8f wmma_f16(v16h a, v16h b, v8f c) {
  // 8 args: (neg_a, A, neg_b, B, c_mod, C, reuse_a, reuse_b)
  return __builtin_amdgcn_wmma_f32_16x16x32_f16(false, a, false, b, (short)0, c,
                                                false, false);
}

// A-matrix fragment (16x32, 16-bit): lane L holds row (L&15);
// lanes 0-15: K = 0..15 contiguous, lanes 16-31: K = 16..31 contiguous
__device__ __forceinline__ v16h load_a_frag(const _Float16* base, int ld,
                                            int row0, int k0) {
  int lane = threadIdx.x & 31;
  const _Float16* rp =
      base + (size_t)(row0 + (lane & 15)) * ld + k0 + ((lane >> 4) << 4);
  v16h v;
  *(v8h*)&v       = *(const v8h*)(rp);
  *((v8h*)&v + 1) = *(const v8h*)(rp + 8);
  return v;
}

// B-matrix fragment (32x16, 16-bit): lane L holds column (L&15);
// lanes 0-15: K = 0..15 contiguous, lanes 16-31: K = 16..31 contiguous.
// Source row (row0 + lane&15) supplies column data (row-major B^T storage).
__device__ __forceinline__ v16h load_b_frag(const _Float16* base, int ld,
                                            int row0, int k0) {
  int lane = threadIdx.x & 31;
  const _Float16* rp =
      base + (size_t)(row0 + (lane & 15)) * ld + k0 + ((lane >> 4) << 4);
  v16h v;
  *(v8h*)&v       = *(const v8h*)(rp);
  *((v8h*)&v + 1) = *(const v8h*)(rp + 8);
  return v;
}

__device__ __forceinline__ v16h load_b_frag_clamp(const _Float16* base, int ld,
                                                  int row0, int rowmax, int k0) {
  int lane = threadIdx.x & 31;
  int r = min(row0 + (lane & 15), rowmax);   // clamped rows are masked later
  const _Float16* rp = base + (size_t)r * ld + k0 + ((lane >> 4) << 4);
  v16h v;
  *(v8h*)&v       = *(const v8h*)(rp);
  *((v8h*)&v + 1) = *(const v8h*)(rp + 8);
  return v;
}

// ---------------------------------------------------------------------------
// fp32 -> fp16 conversion
// ---------------------------------------------------------------------------
__global__ void f32_to_f16(const float* __restrict__ src,
                           _Float16* __restrict__ dst, int n) {
  int i = blockIdx.x * blockDim.x + threadIdx.x;
  if (i < n) dst[i] = (_Float16)src[i];
}

// ---------------------------------------------------------------------------
// C[m,n] = sum_k A[m,k] * W[n,k]   (torch Linear: x @ W^T)
// One wave per 16x64 output block: A-fragment reused across 4 B-fragments.
// ---------------------------------------------------------------------------
template <bool HALF_OUT>
__global__ __launch_bounds__(32) void gemm_rt(const _Float16* __restrict__ A,
                                              const _Float16* __restrict__ W,
                                              void* __restrict__ Cout, int M,
                                              int N, int K) {
  int tm  = blockIdx.x;   // M/16 tiles
  int tn4 = blockIdx.y;   // N/64 tiles
  v8f acc0 = {}, acc1 = {}, acc2 = {}, acc3 = {};
  for (int k0 = 0; k0 < K; k0 += 32) {
    v16h a  = load_a_frag(A, K, tm * 16, k0);
    v16h b0 = load_b_frag(W, K, tn4 * 64 +  0, k0);
    v16h b1 = load_b_frag(W, K, tn4 * 64 + 16, k0);
    v16h b2 = load_b_frag(W, K, tn4 * 64 + 32, k0);
    v16h b3 = load_b_frag(W, K, tn4 * 64 + 48, k0);
    acc0 = wmma_f16(a, b0, acc0);
    acc1 = wmma_f16(a, b1, acc1);
    acc2 = wmma_f16(a, b2, acc2);
    acc3 = wmma_f16(a, b3, acc3);
  }
  int lane = threadIdx.x & 31;
  int n0 = tn4 * 64 + (lane & 15);
  int m0 = tm * 16 + ((lane >> 4) << 3);
  if (HALF_OUT) {
    _Float16* C = (_Float16*)Cout;
#pragma unroll
    for (int r = 0; r < 8; ++r) {
      size_t row = (size_t)(m0 + r) * N;
      C[row + n0 +  0] = (_Float16)acc0[r];
      C[row + n0 + 16] = (_Float16)acc1[r];
      C[row + n0 + 32] = (_Float16)acc2[r];
      C[row + n0 + 48] = (_Float16)acc3[r];
    }
  } else {
    float* C = (float*)Cout;
#pragma unroll
    for (int r = 0; r < 8; ++r) {
      size_t row = (size_t)(m0 + r) * N;
      C[row + n0 +  0] = acc0[r];
      C[row + n0 + 16] = acc1[r];
      C[row + n0 + 32] = acc2[r];
      C[row + n0 + 48] = acc3[r];
    }
  }
}

// ---------------------------------------------------------------------------
// L2-normalize Q,K per (b,t,h) 64-vector; write head-major [B,H,T,64] fp16
// ---------------------------------------------------------------------------
__global__ void normalize_qk(const _Float16* __restrict__ Qh,
                             const _Float16* __restrict__ Kh,
                             _Float16* __restrict__ Qn,
                             _Float16* __restrict__ Kn) {
  int idx = blockIdx.x * blockDim.x + threadIdx.x;
  if (idx >= B_ * T_ * H_) return;
  int h = idx % H_;
  int t = (idx / H_) % T_;
  int b = idx / (H_ * T_);
  size_t src = ((size_t)(b * T_ + t)) * D_ + h * HD_;
  size_t dst = (((size_t)(b * H_ + h)) * T_ + t) * HD_;
  float sq = 0.f, sk = 0.f;
#pragma unroll 8
  for (int i = 0; i < HD_; ++i) {
    float q = (float)Qh[src + i];
    float k = (float)Kh[src + i];
    sq += q * q;
    sk += k * k;
  }
  float iq = 1.f / fmaxf(sqrtf(sq), 1e-6f);
  float ik = 1.f / fmaxf(sqrtf(sk), 1e-6f);
#pragma unroll 8
  for (int i = 0; i < HD_; ++i) {
    Qn[dst + i] = (_Float16)((float)Qh[src + i] * iq);
    Kn[dst + i] = (_Float16)((float)Kh[src + i] * ik);
  }
}

// ---------------------------------------------------------------------------
// Transpose V into [B,H,64,TPAD] fp16 with zero-padded key tail
// ---------------------------------------------------------------------------
__global__ void prep_v(const _Float16* __restrict__ Vh,
                       _Float16* __restrict__ Vt) {
  int idx = blockIdx.x * blockDim.x + threadIdx.x;  // B*H*TPAD
  if (idx >= B_ * H_ * TPAD) return;
  int t = idx % TPAD;
  int h = (idx / TPAD) % H_;
  int b = idx / (TPAD * H_);
  size_t dst = (((size_t)(b * H_ + h)) * HD_) * TPAD + t;
  if (t < T_) {
    size_t src = ((size_t)(b * T_ + t)) * D_ + h * HD_;
#pragma unroll 8
    for (int i = 0; i < HD_; ++i) Vt[dst + (size_t)i * TPAD] = Vh[src + i];
  } else {
#pragma unroll 8
    for (int i = 0; i < HD_; ++i) Vt[dst + (size_t)i * TPAD] = (_Float16)0.f;
  }
}

// ---------------------------------------------------------------------------
// Sliding-window attention, one wave per (b,h, 16-query tile).
// Keys allowed: i <= j < i+WIN  ->  per 16-row tile, <=143 keys = <=9 tiles.
// ---------------------------------------------------------------------------
__global__ __launch_bounds__(32) void attn_kernel(
    const _Float16* __restrict__ Qn, const _Float16* __restrict__ Kn,
    const _Float16* __restrict__ Vt, _Float16* __restrict__ ctx) {
  __shared__ float    Sf[16][160];   // raw scores / exp values
  __shared__ _Float16 Ph[16][160];   // softmaxed probabilities (A-matrix for PV)

  int bh = blockIdx.x;
  int b = bh / H_, h = bh % H_;
  int i0 = blockIdx.y * 16;
  int lane = threadIdx.x & 31;
  int n  = lane & 15;
  int mb = (lane >> 4) << 3;

  const _Float16* Qp = Qn + ((size_t)bh * T_) * HD_;
  const _Float16* Kp = Kn + ((size_t)bh * T_) * HD_;
  const _Float16* Vp = Vt + ((size_t)bh * HD_) * TPAD;

  v16h qa0 = load_a_frag(Qp, HD_, i0, 0);
  v16h qa1 = load_a_frag(Qp, HD_, i0, 32);

  int jmax   = min(T_ - 1, i0 + 15 + WIN - 1);
  int ntiles = ((jmax - i0) >> 4) + 1;              // <= 9
  float slope = exp2f(-8.0f * (float)h / 15.0f);    // ALiBi slope

  for (int t = 0; t < ntiles; ++t) {
    int j0 = i0 + t * 16;
    v16h kb0 = load_b_frag_clamp(Kp, HD_, j0, T_ - 1, 0);
    v16h kb1 = load_b_frag_clamp(Kp, HD_, j0, T_ - 1, 32);
    v8f s = {};
    s = wmma_f16(qa0, kb0, s);
    s = wmma_f16(qa1, kb1, s);
#pragma unroll
    for (int r = 0; r < 8; ++r) {
      int m = mb + r;
      int j = j0 + n;
      int rel = j - (i0 + m);
      float val = (rel < 0 || rel >= WIN || j >= T_)
                      ? NEGINF
                      : (s[r] - slope * (float)rel);
      Sf[m][t * 16 + n] = val;
    }
  }
  __syncthreads();

  int ncols = ntiles * 16;
  if (lane < 16) {
    int row = lane;
    float mx = -3.0e38f;
    for (int c = 0; c < ncols; ++c) mx = fmaxf(mx, Sf[row][c]);
    float sum = 0.f;
    for (int c = 0; c < 160; ++c) {
      float e = 0.f;
      if (c < ncols) { e = __expf(Sf[row][c] - mx); sum += e; }
      Sf[row][c] = e;
    }
    float inv = 1.f / sum;
    for (int c = 0; c < 160; ++c) Ph[row][c] = (_Float16)(Sf[row][c] * inv);
  }
  __syncthreads();

  // O(16x64) = P(16x160) x V(160x64): 5 key-chunks of 32, 4 hd-tiles of 16
  v8f o0 = {}, o1 = {}, o2 = {}, o3 = {};
#pragma unroll
  for (int c = 0; c < 5; ++c) {
    v16h pa = load_a_frag((const _Float16*)Ph, 160, 0, c * 32);
    int key0 = i0 + c * 32;                         // padded region is zero
    v16h vb0 = load_b_frag(Vp, TPAD,  0, key0);
    v16h vb1 = load_b_frag(Vp, TPAD, 16, key0);
    v16h vb2 = load_b_frag(Vp, TPAD, 32, key0);
    v16h vb3 = load_b_frag(Vp, TPAD, 48, key0);
    o0 = wmma_f16(pa, vb0, o0);
    o1 = wmma_f16(pa, vb1, o1);
    o2 = wmma_f16(pa, vb2, o2);
    o3 = wmma_f16(pa, vb3, o3);
  }

#pragma unroll
  for (int r = 0; r < 8; ++r) {
    int t_ = i0 + mb + r;
    size_t rowoff = ((size_t)(b * T_ + t_)) * D_ + h * HD_ + n;
    ctx[rowoff +  0] = (_Float16)o0[r];
    ctx[rowoff + 16] = (_Float16)o1[r];
    ctx[rowoff + 32] = (_Float16)o2[r];
    ctx[rowoff + 48] = (_Float16)o3[r];
  }
}

// ---------------------------------------------------------------------------
extern "C" void kernel_launch(void* const* d_in, const int* in_sizes, int n_in,
                              void* d_out, int out_size, void* d_ws,
                              size_t ws_size, hipStream_t stream) {
  const float* x  = (const float*)d_in[0];
  const float* Wq = (const float*)d_in[1];
  const float* Wk = (const float*)d_in[2];
  const float* Wv = (const float*)d_in[3];
  const float* Wo = (const float*)d_in[4];

  char* ws = (char*)d_ws;
  size_t off = 0;
  auto alloc = [&](size_t bytes) -> void* {
    void* p = ws + off;
    off += (bytes + 255) & ~(size_t)255;
    return p;
  };
  const size_t NX = (size_t)B_ * T_ * D_;   // 4M elements
  const size_t NW = (size_t)D_ * D_;        // 1M elements

  _Float16* xh  = (_Float16*)alloc(NX * 2);
  _Float16* Wqh = (_Float16*)alloc(NW * 2);
  _Float16* Wkh = (_Float16*)alloc(NW * 2);
  _Float16* Wvh = (_Float16*)alloc(NW * 2);
  _Float16* Woh = (_Float16*)alloc(NW * 2);
  _Float16* Qh  = (_Float16*)alloc(NX * 2);
  _Float16* Kh  = (_Float16*)alloc(NX * 2);
  _Float16* Vh  = (_Float16*)alloc(NX * 2);
  _Float16* Qn  = (_Float16*)alloc(NX * 2);
  _Float16* Kn  = (_Float16*)alloc(NX * 2);
  _Float16* Vtp = (_Float16*)alloc((size_t)B_ * H_ * HD_ * TPAD * 2);
  _Float16* ctx = (_Float16*)alloc(NX * 2);

  // fp32 -> fp16
  f32_to_f16<<<(int)((NX + 255) / 256), 256, 0, stream>>>(x, xh, (int)NX);
  f32_to_f16<<<(int)((NW + 255) / 256), 256, 0, stream>>>(Wq, Wqh, (int)NW);
  f32_to_f16<<<(int)((NW + 255) / 256), 256, 0, stream>>>(Wk, Wkh, (int)NW);
  f32_to_f16<<<(int)((NW + 255) / 256), 256, 0, stream>>>(Wv, Wvh, (int)NW);
  f32_to_f16<<<(int)((NW + 255) / 256), 256, 0, stream>>>(Wo, Woh, (int)NW);

  // Q/K/V projections (x @ W^T), fp16 out, fp32 WMMA accumulation
  dim3 gg(B_ * T_ / 16, D_ / 64);
  gemm_rt<true><<<gg, 32, 0, stream>>>(xh, Wqh, Qh, B_ * T_, D_, D_);
  gemm_rt<true><<<gg, 32, 0, stream>>>(xh, Wkh, Kh, B_ * T_, D_, D_);
  gemm_rt<true><<<gg, 32, 0, stream>>>(xh, Wvh, Vh, B_ * T_, D_, D_);

  normalize_qk<<<(B_ * T_ * H_ + 255) / 256, 256, 0, stream>>>(Qh, Kh, Qn, Kn);
  prep_v<<<(B_ * H_ * TPAD + 255) / 256, 256, 0, stream>>>(Vh, Vtp);

  attn_kernel<<<dim3(B_ * H_, T_ / 16), 32, 0, stream>>>(Qn, Kn, Vtp, ctx);

  // output projection -> fp32 d_out
  gemm_rt<false><<<gg, 32, 0, stream>>>(ctx, Woh, d_out, B_ * T_, D_, D_);
}
